// QuantizedLinear_41420664602841
// MI455X (gfx1250) — matrix operand order repaired and checked by reading, not compiled, not run on hardware
//
#include <hip/hip_runtime.h>

typedef __attribute__((ext_vector_type(16))) _Float16 v16h;
typedef __attribute__((ext_vector_type(8)))  _Float16 v8h;
typedef __attribute__((ext_vector_type(4)))  _Float16 v4h;
typedef __attribute__((ext_vector_type(8)))  float    v8f;
typedef __attribute__((ext_vector_type(4)))  float    f32x4;
typedef __attribute__((ext_vector_type(4)))  int      i32x4;

#define BM  128   // workgroup tile M
#define BN  64    // workgroup tile N
#define BK  32    // K per WMMA step
#define LDK 40    // padded LDS row stride in halves (80 B = 20 banks, conflict-free)

// y[m,n] = scale * sum_k x[m,k] * W[n,k] + bias[n]
// A = x tile (BM x BK f16), B = W tile stored [n][k] (BN x BK f16)
__global__ __launch_bounds__(256)
void qlin_f16_wmma(const float* __restrict__ x,
                   const int*   __restrict__ w,
                   const float* __restrict__ scale,
                   const float* __restrict__ bias,
                   float*       __restrict__ out,
                   int M, int N, int K)
{
    __shared__ _Float16 As[2][BM * LDK];   // 2 x 10 KB
    __shared__ _Float16 Bs[2][BN * LDK];   // 2 x 5 KB   (30 KB total)

    const int tid  = threadIdx.x;
    const int lane = tid & 31;
    const int wave = tid >> 5;
    const int mw   = (wave & 3) * 32;      // wave M offset inside tile
    const int nw   = (wave >> 2) * 32;     // wave N offset inside tile

    const int mTile = blockIdx.y * BM;
    const int nTile = blockIdx.x * BN;

    const float* xg = x + (size_t)mTile * K;
    const int*   wg = w + (size_t)nTile * K;

    // staging registers for the next K-step (global -> reg -> convert -> LDS)
    f32x4 xr[4];   // 128 rows x 32 f32 = 1024 float4 / 256 threads = 4 each
    i32x4 wr[2];   //  64 rows x 32 i32 =  512 int4   / 256 threads = 2 each

    v8f acc[2][2];
    {
        v8f z = {};
        acc[0][0] = z; acc[0][1] = z; acc[1][0] = z; acc[1][1] = z;
    }

    auto gload = [&](int k0) {
        #pragma unroll
        for (int i = 0; i < 4; ++i) {
            int idx = tid + i * 256;
            int r = idx >> 3, c = idx & 7;           // 8 float4 per 32-float row
            xr[i] = *(const f32x4*)(xg + (size_t)r * K + k0 + c * 4);
        }
        #pragma unroll
        for (int i = 0; i < 2; ++i) {
            int idx = tid + i * 256;
            int r = idx >> 3, c = idx & 7;
            wr[i] = *(const i32x4*)(wg + (size_t)r * K + k0 + c * 4);
        }
    };

    auto stage = [&](int buf) {
        #pragma unroll
        for (int i = 0; i < 4; ++i) {
            int idx = tid + i * 256;
            int r = idx >> 3, c = idx & 7;
            v4h h;
            h.x = (_Float16)xr[i].x; h.y = (_Float16)xr[i].y;
            h.z = (_Float16)xr[i].z; h.w = (_Float16)xr[i].w;
            *(v4h*)(&As[buf][r * LDK + c * 4]) = h;   // 8B aligned (80B row stride)
        }
        #pragma unroll
        for (int i = 0; i < 2; ++i) {
            int idx = tid + i * 256;
            int r = idx >> 3, c = idx & 7;
            v4h h;
            h.x = (_Float16)wr[i].x; h.y = (_Float16)wr[i].y;
            h.z = (_Float16)wr[i].z; h.w = (_Float16)wr[i].w;
            *(v4h*)(&Bs[buf][r * LDK + c * 4]) = h;
        }
    };

    // A fragment per ISA: lanes 0-15 hold M=lane, K-chunks {0..7, 16..23};
    // lanes 16-31 hold K-chunks {8..15, 24..31}.
    auto ldA = [&](const _Float16* base, int mt) -> v16h {
        int m  = mw + mt * 16 + (lane & 15);
        int kb = (lane >> 4) * 8;
        v8h lo = *(const v8h*)(base + m * LDK + kb);        // 16B aligned
        v8h hi = *(const v8h*)(base + m * LDK + kb + 16);
        v16h a;
        #pragma unroll
        for (int i = 0; i < 8; ++i) { a[i] = lo[i]; a[i + 8] = hi[i]; }
        return a;
    };
    // B fragment: lanes 0-15 hold N=lane with K-chunks {0..7, 16..23}; lanes 16-31 {8..15, 24..31}.
    // Bs is stored [n][k], so each lane's two 8-half chunks are contiguous in LDS.
    auto ldB = [&](const _Float16* base, int nt) -> v16h {
        int n  = nw + nt * 16 + (lane & 15);
        int kb = (lane >> 4) * 8;
        v8h lo = *(const v8h*)(base + n * LDK + kb);
        v8h hi = *(const v8h*)(base + n * LDK + kb + 16);
        v16h b;
        #pragma unroll
        for (int i = 0; i < 8; ++i) { b[i] = lo[i]; b[i + 8] = hi[i]; }
        return b;
    };

    const int KT = K / BK;          // 128 steps
    gload(0);
    stage(0);
    __syncthreads();

    for (int kt = 0; kt < KT; ++kt) {
        const int cur = kt & 1;

        if (kt + 1 < KT) gload((kt + 1) * BK);       // overlap next step's loads
        if (kt + 2 < KT) {                           // prefetch weight stream 2 ahead
            int k2 = (kt + 2) * BK;
            #pragma unroll
            for (int i = 0; i < 2; ++i) {
                int idx = tid + i * 256;
                __builtin_prefetch(wg + (size_t)(idx >> 3) * K + k2 + (idx & 7) * 4, 0, 1);
            }
        }

        const _Float16* ab = As[cur];
        const _Float16* bb = Bs[cur];
        v16h a0 = ldA(ab, 0), a1 = ldA(ab, 1);
        v16h b0 = ldB(bb, 0), b1 = ldB(bb, 1);

        acc[0][0] = __builtin_amdgcn_wmma_f32_16x16x32_f16(false, a0, false, b0,
                                                           (short)0, acc[0][0], false, false);
        acc[0][1] = __builtin_amdgcn_wmma_f32_16x16x32_f16(false, a0, false, b1,
                                                           (short)0, acc[0][1], false, false);
        acc[1][0] = __builtin_amdgcn_wmma_f32_16x16x32_f16(false, a1, false, b0,
                                                           (short)0, acc[1][0], false, false);
        acc[1][1] = __builtin_amdgcn_wmma_f32_16x16x32_f16(false, a1, false, b1,
                                                           (short)0, acc[1][1], false, false);

        if (kt + 1 < KT) stage(cur ^ 1);
        __syncthreads();
    }

    // Epilogue: D layout — VGPR r, lanes 0-15: M=r, N=lane; lanes 16-31: M=r+8.
    const float s      = scale[0];
    const int   col    = lane & 15;
    const int   rowSel = lane >> 4;

    #pragma unroll
    for (int nt = 0; nt < 2; ++nt) {
        const int   gn = nTile + nw + nt * 16 + col;
        const float bv = bias[gn];
        #pragma unroll
        for (int mt = 0; mt < 2; ++mt) {
            const int mBase = mTile + mw + mt * 16 + rowSel * 8;
            #pragma unroll
            for (int r = 0; r < 8; ++r) {
                out[(size_t)(mBase + r) * N + gn] = fmaf(s, acc[mt][nt][r], bv);
            }
        }
    }
}

extern "C" void kernel_launch(void* const* d_in, const int* in_sizes, int n_in,
                              void* d_out, int out_size, void* d_ws, size_t ws_size,
                              hipStream_t stream) {
    const float* x     = (const float*)d_in[0];
    const int*   w     = (const int*)d_in[1];
    const float* scale = (const float*)d_in[2];
    const float* bias  = (const float*)d_in[3];
    float*       out   = (float*)d_out;

    const int D_OUT = in_sizes[3];              // 11008
    const int D_IN  = in_sizes[1] / D_OUT;      // 4096
    const int M     = in_sizes[0] / D_IN;       // 8*32 = 256

    // Shapes divide exactly: M=256=2*BM, N=11008=172*BN, K=4096=128*BK.
    dim3 grid(D_OUT / BN, M / BM);
    qlin_f16_wmma<<<grid, 256, 0, stream>>>(x, w, scale, bias, out, M, D_OUT, D_IN);
}
